// DiffusionAngleHead_18786186952975
// MI455X (gfx1250) — compile-verified, matrix operand order checked by reading, not run on hardware
//
#include <hip/hip_runtime.h>
#include <hip/hip_bf16.h>
#include <math.h>

// ---------------------------------------------------------------------------
// DiffusionAngleHead for MI455X (gfx1250, wave32).
// Heavy GEMMs run on v_wmma_f32_16x16x32_bf16 (f32 accumulate); the DDIM loop
// is algebraically restructured so per-step cost is one 1024x512x512 GEMM.
// WMMA kernels use 8 waves/block (4 N-tiles each) so per-SIMD multi-wave
// residency hides the load->wmma wait the compiler insists on emitting.
// ---------------------------------------------------------------------------

typedef __attribute__((ext_vector_type(16))) __bf16 v16bf;
typedef __attribute__((ext_vector_type(8)))  float  v8f;

union BF16x16 { v16bf v; unsigned int u[8]; uint4 q[2]; };

__device__ __forceinline__ unsigned int pkbf(float a, float b) {
  unsigned int ua = __float_as_uint(a), ub = __float_as_uint(b);
  ua = (ua + 0x7FFFu + ((ua >> 16) & 1u)) >> 16;
  ub = (ub + 0x7FFFu + ((ub >> 16) & 1u)) >> 16;
  return (ua & 0xFFFFu) | (ub << 16);
}

__device__ __forceinline__ float gelu(float x) {
  return 0.5f * x * (1.0f + erff(x * 0.70710678118654752440f));
}

#define WMMA_BF16(acc, a, b)                                                   \
  (acc) = __builtin_amdgcn_wmma_f32_16x16x32_bf16(                             \
      false, (a).v, false, (b).v, (short)0, (acc), false, false)

// B operand (K x N weight tile, 32x16) pre-packed so each lane loads 2x b128.
__device__ __forceinline__ BF16x16 load_btile(const unsigned int* Bp, int kc,
                                              int nt, int ntiles, int lane) {
  BF16x16 b;
  const uint4* p =
      (const uint4*)(Bp + ((((size_t)kc * ntiles + nt) * 32 + lane) << 3));
  b.q[0] = p[0];
  b.q[1] = p[1];
  return b;
}

// A operand from fp32 row-major LDS row (row = &H[m*ldk]).
__device__ __forceinline__ BF16x16 build_a_f32(const float* row, int kc, int hh) {
  BF16x16 a;
#pragma unroll
  for (int v = 0; v < 8; ++v) {
    int k = kc * 32 + ((v < 4) ? 2 * v : 8 + 2 * v) + 8 * hh;
    a.u[v] = pkbf(row[k], row[k + 1]);
  }
  return a;
}

// A operand from packed bf16-pair LDS row (row = &P[m*(K/2)]).
__device__ __forceinline__ BF16x16 build_a_pairs(const unsigned int* row,
                                                 int kc, int hh) {
  BF16x16 a;
#pragma unroll
  for (int v = 0; v < 8; ++v)
    a.u[v] = row[kc * 16 + v + 4 * hh + ((v >> 2) << 2)];
  return a;
}

// --------------------------- weight pre-packing ----------------------------
// Packs row-major W[K][N] fp32 into WMMA-B lane layout, bf16 pairs.
__global__ void pack_b_kernel(const float* __restrict__ W,
                              unsigned int* __restrict__ P, int K, int N) {
  int idx = blockIdx.x * 256 + threadIdx.x;
  int total = (K / 2) * N;
  if (idx >= total) return;
  int v = idx & 7;
  int lane = (idx >> 3) & 31;
  int tile = idx >> 8;
  int ntiles = N >> 4;
  int nt = tile % ntiles;
  int kt = tile / ntiles;
  int h = lane >> 4;
  int n = nt * 16 + (lane & 15);
  int k = kt * 32 + ((v < 4) ? 2 * v : 8 + 2 * v) + 8 * h;
  P[idx] = pkbf(W[(size_t)k * N + n], W[(size_t)(k + 1) * N + n]);
}

// ------------------------------- mean over N -------------------------------
__global__ void mean_kernel(const float* __restrict__ dino,
                            float* __restrict__ xf) {
  int d = blockIdx.x * 256 + threadIdx.x;  // 0..767
  int b = blockIdx.y;
  const float* p = dino + (size_t)b * 256 * 768 + d;
  float s0 = 0.f, s1 = 0.f, s2 = 0.f, s3 = 0.f;
#pragma unroll 4
  for (int n = 0; n < 256; n += 4) {
    s0 += p[(size_t)(n + 0) * 768];
    s1 += p[(size_t)(n + 1) * 768];
    s2 += p[(size_t)(n + 2) * 768];
    s3 += p[(size_t)(n + 3) * 768];
  }
  xf[(size_t)b * 768 + d] = (s0 + s1 + s2 + s3) * (1.0f / 256.0f);
}

// --------------------------- soft-argmax + conf ----------------------------
__device__ __forceinline__ float blk_reduce(float v, float* red, int tid,
                                            int do_max) {
  red[tid] = v;
  __syncthreads();
  for (int s = 128; s > 0; s >>= 1) {
    if (tid < s)
      red[tid] = do_max ? fmaxf(red[tid], red[tid + s]) : (red[tid] + red[tid + s]);
    __syncthreads();
  }
  float r = red[0];
  __syncthreads();
  return r;
}

__global__ void heatmap_kernel(const float* __restrict__ hm,
                               float* __restrict__ out_uv,
                               float* __restrict__ conf) {
  __shared__ float sh[4096];
  __shared__ float red[256];
  int tid = threadIdx.x;
  size_t base = (size_t)blockIdx.x * 4096;
  float mx = -3.402823466e38f;
  for (int i = tid; i < 4096; i += 256) {
    float v = hm[base + i];
    sh[i] = v;
    mx = fmaxf(mx, v);
  }
  float m = blk_reduce(mx, red, tid, 1);
  float se = 0.f, su = 0.f, sv = 0.f;
  for (int i = tid; i < 4096; i += 256) {
    float e = expf(100.0f * (sh[i] - m));
    int w = i & 63, h = i >> 6;
    se += e;
    su += e * (w * (1.0f / 63.0f));
    sv += e * (h * (1.0f / 63.0f));
  }
  float SE = blk_reduce(se, red, tid, 0);
  float SU = blk_reduce(su, red, tid, 0);
  float SV = blk_reduce(sv, red, tid, 0);
  if (tid == 0) {
    out_uv[blockIdx.x * 2 + 0] = SU / SE;
    out_uv[blockIdx.x * 2 + 1] = SV / SE;
    conf[blockIdx.x] = m;
  }
}

// --------------------------- feat encoder (WMMA) ---------------------------
// 768 -> 512 gelu -> 256, writes cond[:,0:256]. 64 blocks x 256 thr (8 waves).
__global__ void feat_mlp_kernel(const float* __restrict__ xf,
                                const unsigned int* __restrict__ W1p,
                                const float* __restrict__ b1,
                                const unsigned int* __restrict__ W2p,
                                const float* __restrict__ b2,
                                float* __restrict__ cond) {
  __shared__ unsigned int XsU[16 * 384];  // xf rows as bf16 pairs
  __shared__ float Hs[16 * 512];
  int tid = threadIdx.x, lane = tid & 31, w = tid >> 5;
  int row0 = blockIdx.x * 16;
  for (int i = tid; i < 16 * 384; i += 256) {
    int m = i / 384, kk = i % 384;
    const float* xp = xf + (size_t)(row0 + m) * 768 + 2 * kk;
    XsU[i] = pkbf(xp[0], xp[1]);
  }
  __syncthreads();

  int m = lane & 15, hh = lane >> 4;
  // layer 1: N=512 (32 tiles, 4/wave), K=768 (24 chunks)
  v8f acc[4];
#pragma unroll
  for (int t = 0; t < 4; ++t)
#pragma unroll
    for (int r = 0; r < 8; ++r) acc[t][r] = 0.0f;
  for (int kc = 0; kc < 24; ++kc) {
    BF16x16 bb[4];
#pragma unroll
    for (int t = 0; t < 4; ++t) bb[t] = load_btile(W1p, kc, w * 4 + t, 32, lane);
    BF16x16 a = build_a_pairs(&XsU[m * 384], kc, hh);
#pragma unroll
    for (int t = 0; t < 4; ++t) WMMA_BF16(acc[t], a, bb[t]);
  }
#pragma unroll
  for (int t = 0; t < 4; ++t) {
    int n = (w * 4 + t) * 16 + (lane & 15);
    float bia = b1[n];
#pragma unroll
    for (int r = 0; r < 8; ++r) Hs[(r + 8 * hh) * 512 + n] = gelu(acc[t][r] + bia);
  }
  __syncthreads();

  // layer 2: N=256 (16 tiles, 2/wave), K=512 (16 chunks)
  v8f ac2[2];
#pragma unroll
  for (int t = 0; t < 2; ++t)
#pragma unroll
    for (int r = 0; r < 8; ++r) ac2[t][r] = 0.0f;
  for (int kc = 0; kc < 16; ++kc) {
    BF16x16 bb[2];
#pragma unroll
    for (int t = 0; t < 2; ++t) bb[t] = load_btile(W2p, kc, w * 2 + t, 16, lane);
    BF16x16 a = build_a_f32(&Hs[m * 512], kc, hh);
#pragma unroll
    for (int t = 0; t < 2; ++t) WMMA_BF16(ac2[t], a, bb[t]);
  }
#pragma unroll
  for (int t = 0; t < 2; ++t) {
    int n = (w * 2 + t) * 16 + (lane & 15);
    float bia = b2[n];
#pragma unroll
    for (int r = 0; r < 8; ++r)
      cond[(size_t)(row0 + r + 8 * hh) * 512 + n] = ac2[t][r] + bia;
  }
}

// ------------------------- uv encoder + conf gate --------------------------
__global__ void uvgate_kernel(const float* __restrict__ uv,
                              const float* __restrict__ conf,
                              const float* __restrict__ uw1,
                              const float* __restrict__ ub1,
                              const float* __restrict__ uw2,
                              const float* __restrict__ ub2,
                              const float* __restrict__ cw1,
                              const float* __restrict__ cb1,
                              const float* __restrict__ cw2,
                              const float* __restrict__ cb2,
                              float* __restrict__ cond) {
  __shared__ float uvl[14], cl[7], h1[256], hc[128], gsh;
  int b = blockIdx.x, tid = threadIdx.x;
  if (tid < 14) uvl[tid] = uv[b * 14 + tid];
  if (tid >= 32 && tid < 39) cl[tid - 32] = conf[b * 7 + (tid - 32)];
  __syncthreads();
  {
    float s = ub1[tid];
#pragma unroll
    for (int i = 0; i < 14; ++i) s += uvl[i] * uw1[i * 256 + tid];
    h1[tid] = gelu(s);
  }
  if (tid < 128) {
    float s = cb1[tid];
#pragma unroll
    for (int i = 0; i < 7; ++i) s += cl[i] * cw1[i * 128 + tid];
    hc[tid] = gelu(s);
  }
  __syncthreads();
  if (tid == 0) {
    float s = cb2[0];
    for (int i = 0; i < 128; ++i) s += hc[i] * cw2[i];
    gsh = 1.0f / (1.0f + expf(-s));
  }
  __syncthreads();
  float s = ub2[tid];
  for (int i = 0; i < 256; ++i) s += h1[i] * uw2[i * 256 + tid];
  cond[(size_t)b * 512 + 256 + tid] = s * gsh;
}

// ---------------------- init head: 512 -> 256 gelu -> 6 --------------------
__global__ void init_mlp_kernel(const float* __restrict__ cond,
                                const unsigned int* __restrict__ W1p,
                                const float* __restrict__ b1,
                                const float* __restrict__ w2,
                                const float* __restrict__ b2,
                                float* __restrict__ xbuf,
                                float* __restrict__ out_init) {
  __shared__ unsigned int Cs[16 * 256];
  __shared__ float Hs[16 * 256];
  int tid = threadIdx.x, lane = tid & 31, w = tid >> 5;
  int row0 = blockIdx.x * 16;
  for (int i = tid; i < 16 * 256; i += 256) {
    int m = i / 256, kk = i % 256;
    const float* cp = cond + (size_t)(row0 + m) * 512 + 2 * kk;
    Cs[i] = pkbf(cp[0], cp[1]);
  }
  __syncthreads();
  int m = lane & 15, hh = lane >> 4;
  // N=256 (16 tiles, 2/wave), K=512 (16 chunks)
  v8f acc[2];
#pragma unroll
  for (int t = 0; t < 2; ++t)
#pragma unroll
    for (int r = 0; r < 8; ++r) acc[t][r] = 0.0f;
  for (int kc = 0; kc < 16; ++kc) {
    BF16x16 bb[2];
#pragma unroll
    for (int t = 0; t < 2; ++t) bb[t] = load_btile(W1p, kc, w * 2 + t, 16, lane);
    BF16x16 a = build_a_pairs(&Cs[m * 256], kc, hh);
#pragma unroll
    for (int t = 0; t < 2; ++t) WMMA_BF16(acc[t], a, bb[t]);
  }
#pragma unroll
  for (int t = 0; t < 2; ++t) {
    int n = (w * 2 + t) * 16 + (lane & 15);
    float bia = b1[n];
#pragma unroll
    for (int r = 0; r < 8; ++r) Hs[(r + 8 * hh) * 256 + n] = gelu(acc[t][r] + bia);
  }
  __syncthreads();
  if (tid < 96) {
    int mm = tid / 6, c = tid % 6;
    float s = b2[c];
    for (int i = 0; i < 256; ++i) s += Hs[mm * 256 + i] * w2[i * 6 + c];
    int row = row0 + mm;
    out_init[row * 6 + c] = s;
    xbuf[row * 6 + c] = s;
  }
}

// --------------- C1 = cond @ dn_w1[134:646] + b1  (hoisted) ----------------
__global__ void c1_kernel(const float* __restrict__ cond,
                          const unsigned int* __restrict__ Wp,
                          const float* __restrict__ b1,
                          float* __restrict__ C1) {
  __shared__ unsigned int Cs[16 * 256];
  int tid = threadIdx.x, lane = tid & 31, w = tid >> 5;
  int row0 = blockIdx.x * 16;
  for (int i = tid; i < 16 * 256; i += 256) {
    int m = i / 256, kk = i % 256;
    const float* cp = cond + (size_t)(row0 + m) * 512 + 2 * kk;
    Cs[i] = pkbf(cp[0], cp[1]);
  }
  __syncthreads();
  int m = lane & 15, hh = lane >> 4;
  // N=512 (32 tiles, 4/wave), K=512 (16 chunks)
  v8f acc[4];
#pragma unroll
  for (int t = 0; t < 4; ++t)
#pragma unroll
    for (int r = 0; r < 8; ++r) acc[t][r] = 0.0f;
  for (int kc = 0; kc < 16; ++kc) {
    BF16x16 bb[4];
#pragma unroll
    for (int t = 0; t < 4; ++t) bb[t] = load_btile(Wp, kc, w * 4 + t, 32, lane);
    BF16x16 a = build_a_pairs(&Cs[m * 256], kc, hh);
#pragma unroll
    for (int t = 0; t < 4; ++t) WMMA_BF16(acc[t], a, bb[t]);
  }
#pragma unroll
  for (int t = 0; t < 4; ++t) {
    int n = (w * 4 + t) * 16 + (lane & 15);
    float bia = b1[n];
#pragma unroll
    for (int r = 0; r < 8; ++r)
      C1[(size_t)(row0 + r + 8 * hh) * 512 + n] = acc[t][r] + bia;
  }
}

// ---- per-step: tvec[512] = time_mlp(sin-emb(t)) @ dn_w1[6:134] ------------
__global__ void temb_kernel(float tval, const float* __restrict__ tw1,
                            const float* __restrict__ tb1,
                            const float* __restrict__ tw2,
                            const float* __restrict__ tb2,
                            const float* __restrict__ dnw1,
                            float* __restrict__ tvec) {
  __shared__ float emb[128], h[512], te[128];
  int tid = threadIdx.x;
  if (tid < 64) {
    float f = expf(-9.2103403719761836f * (float)tid / 63.0f);
    float a = tval * f;
    emb[tid] = sinf(a);
    emb[64 + tid] = cosf(a);
  }
  __syncthreads();
  {
    float s = tb1[tid];
    for (int i = 0; i < 128; ++i) s += emb[i] * tw1[i * 512 + tid];
    h[tid] = gelu(s);
  }
  __syncthreads();
  if (tid < 128) {
    float s = tb2[tid];
    for (int i = 0; i < 512; ++i) s += h[i] * tw2[i * 128 + tid];
    te[tid] = s;
  }
  __syncthreads();
  {
    float s = 0.f;
    for (int i = 0; i < 128; ++i) s += te[i] * dnw1[(size_t)(6 + i) * 512 + tid];
    tvec[tid] = s;
  }
}

// ------------------------- fused DDIM denoise step -------------------------
__global__ void step_kernel(const float* __restrict__ C1,
                            const float* __restrict__ tvec,
                            const float* __restrict__ dnw1,  // rows 0..5
                            const unsigned int* __restrict__ W2p,
                            const float* __restrict__ b2,
                            const float* __restrict__ w3,
                            const float* __restrict__ b3,
                            float* __restrict__ xbuf,
                            float* __restrict__ out_ang, int last, float s1,
                            float inv_sab, float s3, float s4) {
  __shared__ float Xs[96];
  __shared__ float W1x[6 * 512];
  __shared__ float Tv[512];
  __shared__ float Hs[16 * 512];
  int tid = threadIdx.x, lane = tid & 31, w = tid >> 5;
  int row0 = blockIdx.x * 16;
  if (tid < 96) Xs[tid] = xbuf[row0 * 6 + tid];
  for (int i = tid; i < 3072; i += 256) W1x[i] = dnw1[i];
  for (int i = tid; i < 512; i += 256) Tv[i] = tvec[i];
  __syncthreads();

  // h1 = gelu(C1 + tvec + x @ W1x)
  for (int idx = tid; idx < 16 * 512; idx += 256) {
    int m = idx >> 9, j = idx & 511;
    float v = C1[(size_t)(row0 + m) * 512 + j] + Tv[j];
#pragma unroll
    for (int i = 0; i < 6; ++i) v += Xs[m * 6 + i] * W1x[i * 512 + j];
    Hs[m * 512 + j] = gelu(v);
  }
  __syncthreads();

  // h2 = gelu(h1 @ dn_w2 + b2) via WMMA; accs held in regs across the barrier
  int m = lane & 15, hh = lane >> 4;
  // N=512 (32 tiles, 4/wave), K=512 (16 chunks)
  v8f acc[4];
#pragma unroll
  for (int t = 0; t < 4; ++t)
#pragma unroll
    for (int r = 0; r < 8; ++r) acc[t][r] = 0.0f;
  for (int kc = 0; kc < 16; ++kc) {
    BF16x16 bb[4];
#pragma unroll
    for (int t = 0; t < 4; ++t) bb[t] = load_btile(W2p, kc, w * 4 + t, 32, lane);
    BF16x16 a = build_a_f32(&Hs[m * 512], kc, hh);
#pragma unroll
    for (int t = 0; t < 4; ++t) WMMA_BF16(acc[t], a, bb[t]);
  }
  __syncthreads();  // all waves done reading Hs (h1) -> safe to overwrite
#pragma unroll
  for (int t = 0; t < 4; ++t) {
    int n = (w * 4 + t) * 16 + (lane & 15);
    float bia = b2[n];
#pragma unroll
    for (int r = 0; r < 8; ++r) Hs[(r + 8 * hh) * 512 + n] = gelu(acc[t][r] + bia);
  }
  __syncthreads();

  // noise_pred + DDIM update
  if (tid < 96) {
    int mm = tid / 6, c = tid % 6;
    float np = b3[c];
    for (int i = 0; i < 512; ++i) np += Hs[mm * 512 + i] * w3[i * 6 + c];
    float xv = Xs[tid];
    float px0 = (xv - s1 * np) * inv_sab;
    px0 = fminf(fmaxf(px0, -4.0f), 4.0f);
    float xn = s3 * px0 + s4 * np;
    xbuf[row0 * 6 + tid] = xn;
    if (last) out_ang[row0 * 6 + tid] = xn;
  }
}

// ---------------------------------------------------------------------------
extern "C" void kernel_launch(void* const* d_in, const int* in_sizes, int n_in,
                              void* d_out, int out_size, void* d_ws,
                              size_t ws_size, hipStream_t stream) {
  const float* dino    = (const float*)d_in[0];
  const float* hm      = (const float*)d_in[1];
  const float* time_w1 = (const float*)d_in[2];
  const float* time_b1 = (const float*)d_in[3];
  const float* time_w2 = (const float*)d_in[4];
  const float* time_b2 = (const float*)d_in[5];
  const float* uv_w1   = (const float*)d_in[6];
  const float* uv_b1   = (const float*)d_in[7];
  const float* uv_w2   = (const float*)d_in[8];
  const float* uv_b2   = (const float*)d_in[9];
  const float* conf_w1 = (const float*)d_in[10];
  const float* conf_b1 = (const float*)d_in[11];
  const float* conf_w2 = (const float*)d_in[12];
  const float* conf_b2 = (const float*)d_in[13];
  const float* feat_w1 = (const float*)d_in[14];
  const float* feat_b1 = (const float*)d_in[15];
  const float* feat_w2 = (const float*)d_in[16];
  const float* feat_b2 = (const float*)d_in[17];
  const float* dn_w1   = (const float*)d_in[18];
  const float* dn_b1   = (const float*)d_in[19];
  const float* dn_w2   = (const float*)d_in[20];
  const float* dn_b2   = (const float*)d_in[21];
  const float* dn_w3   = (const float*)d_in[22];
  const float* dn_b3   = (const float*)d_in[23];
  const float* init_w1 = (const float*)d_in[24];
  const float* init_b1 = (const float*)d_in[25];
  const float* init_w2 = (const float*)d_in[26];
  const float* init_b2 = (const float*)d_in[27];

  float* out      = (float*)d_out;
  float* out_uv   = out + 1024 * 6;
  float* out_init = out + 1024 * 6 + 1024 * 14;

  // workspace layout (fp32 slots)
  float* ws = (float*)d_ws;
  size_t off = 0;
  float* xf   = ws + off; off += (size_t)1024 * 768;
  float* conf = ws + off; off += 1024 * 7;
  float* cond = ws + off; off += (size_t)1024 * 512;
  float* C1   = ws + off; off += (size_t)1024 * 512;
  float* tvec = ws + off; off += 512;
  float* xbuf = ws + off; off += 1024 * 6;
  unsigned int* fw1p = (unsigned int*)(ws + off); off += 768 * 512 / 2;
  unsigned int* fw2p = (unsigned int*)(ws + off); off += 512 * 256 / 2;
  unsigned int* iw1p = (unsigned int*)(ws + off); off += 512 * 256 / 2;
  unsigned int* dw1cp = (unsigned int*)(ws + off); off += 512 * 512 / 2;
  unsigned int* dw2p  = (unsigned int*)(ws + off); off += 512 * 512 / 2;

  auto pack = [&](const float* W, unsigned int* P, int K, int N) {
    int total = K * N / 2;
    pack_b_kernel<<<(total + 255) / 256, 256, 0, stream>>>(W, P, K, N);
  };

  mean_kernel<<<dim3(3, 1024), 256, 0, stream>>>(dino, xf);
  heatmap_kernel<<<1024 * 7, 256, 0, stream>>>(hm, out_uv, conf);

  pack(feat_w1, fw1p, 768, 512);
  pack(feat_w2, fw2p, 512, 256);
  pack(init_w1, iw1p, 512, 256);
  pack(dn_w1 + (size_t)134 * 512, dw1cp, 512, 512);
  pack(dn_w2, dw2p, 512, 512);

  feat_mlp_kernel<<<64, 256, 0, stream>>>(xf, fw1p, feat_b1, fw2p, feat_b2, cond);
  uvgate_kernel<<<1024, 256, 0, stream>>>(out_uv, conf, uv_w1, uv_b1, uv_w2,
                                          uv_b2, conf_w1, conf_b1, conf_w2,
                                          conf_b2, cond);
  init_mlp_kernel<<<64, 256, 0, stream>>>(cond, iw1p, init_b1, init_w2, init_b2,
                                          xbuf, out_init);
  c1_kernel<<<64, 256, 0, stream>>>(cond, dw1cp, dn_b1, C1);

  // alphas_cumprod (host-side pure math; baked into graph as launch args)
  float ac[20];
  double prod = 1.0;
  for (int i = 0; i < 20; ++i) {
    double beta = 1e-4 + (0.02 - 1e-4) * (double)i / 19.0;
    prod *= (1.0 - beta);
    ac[i] = (float)prod;
  }

  for (int t = 19; t >= 0; --t) {
    temb_kernel<<<1, 512, 0, stream>>>((float)t, time_w1, time_b1, time_w2,
                                       time_b2, dn_w1, tvec);
    float abt = ac[t];
    float abp = (t > 0) ? ac[t - 1] : 1.0f;
    step_kernel<<<64, 256, 0, stream>>>(
        C1, tvec, dn_w1, dw2p, dn_b2, dn_w3, dn_b3, xbuf, out, (t == 0) ? 1 : 0,
        sqrtf(1.0f - abt), 1.0f / sqrtf(abt), sqrtf(abp), sqrtf(1.0f - abp));
  }
}
